// PAM_8186207666342
// MI455X (gfx1250) — matrix operand order repaired and checked by reading, not compiled, not run on hardware
//
#include <hip/hip_runtime.h>
#include <hip/hip_bf16.h>
#include <stdint.h>

// ---------------------------------------------------------------------------
// PAM (parallax attention) for MI455X / gfx1250, wave32 + WMMA f16.
// ---------------------------------------------------------------------------

typedef _Float16 half_t;
typedef __attribute__((ext_vector_type(16))) _Float16 v16h;
typedef __attribute__((ext_vector_type(8)))  float    v8f;

#define BB 4
#define CC 64
#define C2 128
#define HT 128
#define WD 256
#define HW (HT*WD)            // 32768

// ---------------- WMMA wrapper (probe-verified signature) ------------------
__device__ __forceinline__ v8f wmma16(v16h a, v16h b, v8f c) {
  return __builtin_amdgcn_wmma_f32_16x16x32_f16(false, a, false, b, (short)0, c,
                                                false, false);
}

// ---------------- async global->LDS (b128) via inline asm ------------------
// ISA: VDST = LDS byte address VGPR, VADDR = 64-bit global address, "off" =
// no SADDR. Tracked by ASYNCcnt.
__device__ __forceinline__ void g2l_b128(void* lds, const void* gptr) {
  unsigned           lo = (unsigned)(size_t)lds;            // LDS byte offset
  unsigned long long ga = (unsigned long long)(size_t)gptr; // global address
  asm volatile("global_load_async_to_lds_b128 %0, %1, off"
               :: "v"(lo), "v"(ga)
               : "memory");
}

__device__ __forceinline__ void wait_async() {
  asm volatile("s_wait_asynccnt 0" ::: "memory");
}

// ===========================================================================
// K1: cat = concat(conv1x1(x)+bias, x), then BN -> f16 (NCHW, 128 ch)
// One wave per 16-pixel tile: WMMA (M=16 px, N=64 oc, K=64 ic).
// ===========================================================================
__global__ __launch_bounds__(128) void pam_front(
    const float* __restrict__ x, const float* __restrict__ w1,
    const float* __restrict__ bia,
    const float* __restrict__ bng, const float* __restrict__ bnb,
    const float* __restrict__ bnm, const float* __restrict__ bnv,
    half_t* __restrict__ cat)
{
  const int tid = threadIdx.x, lane = tid & 31, wv = tid >> 5;
  const int t  = blockIdx.x * 4 + wv;          // 8192 wave-tiles
  const int wt = t & 15, h = (t >> 4) & 127, b = t >> 11;
  const int w0 = wt * 16;
  const int m = lane & 15, kh = lane >> 4;
  const float* xb = x + ((size_t)b*CC*HT + h)*WD + w0;   // + k*HW + m

  // A fragments: 16 px x 64 ic (2 chunks of K=32)
  v16h a[2];
#pragma unroll
  for (int kc = 0; kc < 2; ++kc) {
#pragma unroll
    for (int e = 0; e < 16; ++e) {
      int k = kc*32 + ((e < 8) ? (kh*8 + e) : (16 + kh*8 + (e - 8)));
      a[kc][e] = (half_t)xb[(size_t)k*HW + m];
    }
  }

#pragma unroll
  for (int nt = 0; nt < 4; ++nt) {
    v8f acc = {0.f,0.f,0.f,0.f,0.f,0.f,0.f,0.f};
#pragma unroll
    for (int kc = 0; kc < 2; ++kc) {
      v16h bf;
#pragma unroll
      for (int e = 0; e < 16; ++e) {
        int k = kc*32 + ((e < 8) ? (kh*8 + e) : (16 + kh*8 + (e - 8)));
        bf[e] = (half_t)w1[(nt*16 + (lane & 15))*64 + k];    // B[k][n] = W[n][k]
      }
      acc = wmma16(a[kc], bf, acc);
    }
    const int oc  = nt*16 + (lane & 15);
    const float g = bng[oc], bt = bnb[oc], mu = bnm[oc];
    const float inv = rsqrtf(bnv[oc] + 1e-5f);
    half_t* outp = cat + ((size_t)(b*C2 + oc)*HT + h)*WD + w0;
#pragma unroll
    for (int r = 0; r < 8; ++r) {
      int row = r + 8*kh;
      float v = acc[r] + bia[oc];
      outp[row] = (half_t)(g * (v - mu) * inv + bt);
    }
  }

  // channels 64..127: BN(x) passthrough
  for (int idx = lane; idx < 16*64; idx += 32) {
    int mm = idx & 15, k = idx >> 4, oc = 64 + k;
    float v = xb[(size_t)k*HW + mm];
    float inv = rsqrtf(bnv[oc] + 1e-5f);
    v = bng[oc] * (v - bnm[oc]) * inv + bnb[oc];
    cat[((size_t)(b*C2 + oc)*HT + h)*WD + w0 + mm] = (half_t)v;
  }
}

// ===========================================================================
// K2: grouped 3x3 conv (groups=4, 128->128, pad=1), implicit GEMM.
// Block: (b, g, h, 64-px strip). LDS: input tile [3][66][32] + weights
// pre-swizzled into exact B-fragment order (one 32B contiguous read / lane).
// ===========================================================================
__global__ __launch_bounds__(128) void pam_conv3x3g4(
    const half_t* __restrict__ in, const half_t* __restrict__ res,
    half_t* __restrict__ out,
    const float* __restrict__ wc, const float* __restrict__ bias, int do_lrelu)
{
  __shared__ half_t inT[3*66*32];        // [row][x][ic]
  __shared__ half_t wbt[9*2*32*16];      // [tap][ntile][lane][e]  (B-swizzled)
  const int tid = threadIdx.x;
  const int wq = blockIdx.x & 3;
  const int h0 = (blockIdx.x >> 2) & 127;
  const int g  = (blockIdx.x >> 9) & 3;
  const int b  = blockIdx.x >> 11;
  const int w0 = wq * 64;

  // weights -> LDS, pre-swizzled for B fragments
  for (int lin = tid; lin < 9*2*32*16; lin += 128) {
    int e = lin & 15, ln = (lin >> 4) & 31, nt = (lin >> 9) & 1, tap = lin >> 10;
    int k = (e < 8) ? ((ln >> 4)*8 + e) : (16 + (ln >> 4)*8 + (e - 8));   // ic within group
    int n = nt*16 + (ln & 15);           // oc within group
    wbt[lin] = (half_t)wc[(((g*32 + n)*32) + k)*9 + tap];
  }
  // input halo tile -> LDS (zero padded)
  for (int lin = tid; lin < 3*66*32; lin += 128) {
    int x = lin % 66, icr = lin / 66, ic = icr & 31, row = icr >> 5;
    int hh = h0 - 1 + row, xg = w0 - 1 + x;
    half_t v = (half_t)0.f;
    if (hh >= 0 && hh < HT && xg >= 0 && xg < WD)
      v = in[((size_t)(b*C2 + g*32 + ic)*HT + hh)*WD + xg];
    inT[(row*66 + x)*32 + ic] = v;
  }
  __syncthreads();

  const int lane = tid & 31, mt = tid >> 5;       // 4 waves = 4 M-tiles
  const int m = lane & 15, kh = lane >> 4;
  v8f z = {0.f,0.f,0.f,0.f,0.f,0.f,0.f,0.f};
  v8f acc[2] = {z, z};
#pragma unroll
  for (int tap = 0; tap < 9; ++tap) {
    int dy = tap / 3, dx = tap % 3;
    const half_t* ap = &inT[(dy*66 + mt*16 + dx + m)*32];
    v16h a;
#pragma unroll
    for (int e = 0; e < 8; ++e) { a[e] = ap[kh*8 + e]; a[8+e] = ap[16 + kh*8 + e]; }
#pragma unroll
    for (int nt = 0; nt < 2; ++nt) {
      const half_t* bp = &wbt[((tap*2 + nt)*32 + lane)*16];
      v16h bf;
#pragma unroll
      for (int e = 0; e < 16; ++e) bf[e] = bp[e];
      acc[nt] = wmma16(a, bf, acc[nt]);
    }
  }
#pragma unroll
  for (int nt = 0; nt < 2; ++nt) {
    int oc = g*32 + nt*16 + (lane & 15);
    float bs = bias[oc];
    size_t base = ((size_t)(b*C2 + oc)*HT + h0)*WD + w0 + mt*16;
#pragma unroll
    for (int r = 0; r < 8; ++r) {
      int row = r + 8*kh;
      float v = acc[nt][r] + bs;
      if (do_lrelu) v = (v > 0.f) ? v : 0.1f*v;
      if (res) v += (float)res[base + row];
      out[base + row] = (half_t)v;
    }
  }
}

// ===========================================================================
// K3: 1x1 conv, groups=2 (128 -> 64), f16 in -> f32 out (Q or K features)
// ===========================================================================
__global__ __launch_bounds__(128) void pam_proj1x1g2(
    const half_t* __restrict__ in, const float* __restrict__ wt,
    const float* __restrict__ bias, float* __restrict__ out)
{
  const int tid = threadIdx.x, lane = tid & 31, wv = tid >> 5;
  const int t  = blockIdx.x * 4 + wv;          // 16384 wave-tiles
  const int wtile = t & 15, h = (t >> 4) & 127, g = (t >> 11) & 1, b = t >> 12;
  const int w0 = wtile * 16;
  const int m = lane & 15, kh = lane >> 4;
  const half_t* ib = in + ((size_t)(b*C2 + g*64)*HT + h)*WD + w0;

  v16h a[2];
#pragma unroll
  for (int kc = 0; kc < 2; ++kc) {
#pragma unroll
    for (int e = 0; e < 16; ++e) {
      int k = kc*32 + ((e < 8) ? (kh*8 + e) : (16 + kh*8 + (e - 8)));
      a[kc][e] = ib[(size_t)k*HW + m];
    }
  }
#pragma unroll
  for (int nt = 0; nt < 2; ++nt) {
    v8f acc = {0.f,0.f,0.f,0.f,0.f,0.f,0.f,0.f};
#pragma unroll
    for (int kc = 0; kc < 2; ++kc) {
      v16h bf;
#pragma unroll
      for (int e = 0; e < 16; ++e) {
        int k = kc*32 + ((e < 8) ? (kh*8 + e) : (16 + kh*8 + (e - 8)));
        bf[e] = (half_t)wt[(g*32 + nt*16 + (lane & 15))*64 + k];
      }
      acc = wmma16(a[kc], bf, acc);
    }
    int oc = g*32 + nt*16 + (lane & 15);
    size_t base = ((size_t)(b*CC + oc)*HT + h)*WD + w0;
    float bs = bias[oc];
#pragma unroll
    for (int r = 0; r < 8; ++r) out[base + r + 8*kh] = acc[r] + bs;
  }
}

// ===========================================================================
// K4: fused attention per (b,h) slice. LDS layout (dynamic, 270336 B):
//   s     f16[256][256]  logits            @      0 (131072)
//   bufA  f16[256][64]   Qhat, later xl    @ 131072 ( 32768)
//   bufB  f16            Khat[c][w], later xr[w][c] @ 163840 ( 32768)
//   stats f32 rmax/rsum/cmax/csum[256]     @ 196608 (  4096)
//   Vl,Vr,mean                             @ 200704 (  4096)
//   stage f32[64][256]  async landing pad  @ 204800 ( 65536)
// ===========================================================================
#define ATTN_LDS_BYTES 270336

__device__ __forceinline__ void stage64x256(const float* src, float* stage, int tid) {
  for (int i = tid; i < 64*64; i += 256) {     // 4096 x b128
    int c = i >> 6, seg = i & 63;
    g2l_b128(stage + c*WD + seg*4, src + (size_t)c*HW + seg*4);
  }
}

__global__ __launch_bounds__(256) void pam_attn(
    const float* __restrict__ Qg, const float* __restrict__ Kg,
    const float* __restrict__ xl, const float* __restrict__ xr,
    float* __restrict__ outl, float* __restrict__ outr)
{
  extern __shared__ char smem[];
  half_t* s    = (half_t*)(smem);
  half_t* bufA = (half_t*)(smem + 131072);
  half_t* bufB = (half_t*)(smem + 163840);
  float* rmax  = (float*)(smem + 196608);
  float* rsum  = rmax + 256;
  float* cmax  = rsum + 256;
  float* csum  = cmax + 256;
  float* Vl    = (float*)(smem + 200704);
  float* Vr    = Vl + 256;
  float* mean  = Vr + 256;
  float* stage = (float*)(smem + 204800);

  const int tid = threadIdx.x, lane = tid & 31, wv = tid >> 5;
  const int n = blockIdx.x, b = n >> 7, h = n & 127;
  const size_t rowbase = ((size_t)b*CC*HT + h)*WD;
  const int m = lane & 15, kh = lane >> 4;

  // ---- Qhat: async load, per-channel mean subtract, f16 [w][c] ----
  stage64x256(Qg + rowbase, stage, tid);
  wait_async(); __syncthreads();
  if (tid < 64) { float sm = 0.f;
    for (int w2 = 0; w2 < WD; ++w2) sm += stage[tid*WD + w2];
    mean[tid] = sm * (1.f/WD); }
  __syncthreads();
  for (int i = tid; i < 64*WD; i += 256) {
    int c = i >> 8, w2 = i & 255;
    bufA[w2*64 + c] = (half_t)(stage[c*WD + w2] - mean[c]);
  }
  __syncthreads();
  // ---- Khat: f16 [c][w] (B-major for score GEMM) ----
  stage64x256(Kg + rowbase, stage, tid);
  wait_async(); __syncthreads();
  if (tid < 64) { float sm = 0.f;
    for (int w2 = 0; w2 < WD; ++w2) sm += stage[tid*WD + w2];
    mean[tid] = sm * (1.f/WD); }
  __syncthreads();
  for (int i = tid; i < 64*WD; i += 256) {
    int c = i >> 8, w2 = i & 255;
    bufB[c*WD + w2] = (half_t)(stage[c*WD + w2] - mean[c]);
  }
  __syncthreads();

  // ---- score GEMM: s[i][j] = sum_c Qhat[i][c]*Khat[c][j] ----
  for (int t = wv; t < 256; t += 8) {
    int mi = t >> 4, nj = t & 15;
    v8f acc = {0.f,0.f,0.f,0.f,0.f,0.f,0.f,0.f};
#pragma unroll
    for (int kc = 0; kc < 2; ++kc) {
      const half_t* ap = bufA + (mi*16 + m)*64 + kc*32;
      v16h a;
#pragma unroll
      for (int e = 0; e < 8; ++e) { a[e] = ap[kh*8 + e]; a[8+e] = ap[16 + kh*8 + e]; }
      const half_t* bp = bufB + (size_t)(kc*32 + kh*8)*WD + nj*16 + (lane & 15);
      v16h bf;
#pragma unroll
      for (int e = 0; e < 16; ++e) bf[e] = bp[((e < 8) ? e : (8 + e))*WD];
      acc = wmma16(a, bf, acc);
    }
    int col = nj*16 + (lane & 15);
    int rb2 = mi*16 + 8*kh;
#pragma unroll
    for (int r = 0; r < 8; ++r) s[(rb2 + r)*WD + col] = (half_t)acc[r];
  }
  __syncthreads();

  // overlap: start staging x_left while stats are computed
  stage64x256(xl + rowbase, stage, tid);

  // ---- softmax stats (row for M_r2l, col for M_l2r) ----
  {
    const int i = tid;
    float mx = -3.0e38f;
    for (int j = 0; j < WD; ++j) mx = fmaxf(mx, (float)s[i*WD + j]);
    float sm = 0.f;
    for (int j = 0; j < WD; ++j) sm += __expf((float)s[i*WD + j] - mx);
    rmax[i] = mx; rsum[i] = sm;
    float cm = -3.0e38f;
    for (int j = 0; j < WD; ++j) cm = fmaxf(cm, (float)s[j*WD + i]);
    float cs = 0.f;
    for (int j = 0; j < WD; ++j) cs += __expf((float)s[j*WD + i] - cm);
    cmax[i] = cm; csum[i] = cs;
  }
  wait_async(); __syncthreads();
  for (int i = tid; i < 64*WD; i += 256) {           // xl -> bufA [w][c] f16
    int c = i >> 8, w2 = i & 255;
    bufA[w2*64 + c] = (half_t)stage[c*WD + w2];
  }
  __syncthreads();
  stage64x256(xr + rowbase, stage, tid);             // start staging x_right

  // ---- valid-mask einsums (relax window +-2) ----
  {
    const int i = tid;
    float vl = 0.f, vr = 0.f;
    for (int j = 0; j < WD; ++j) {
      float sij = (float)s[i*WD + j];
      float sji = (float)s[j*WD + i];
      float a = 0.f, ar = 0.f;
#pragma unroll
      for (int d = -2; d <= 2; ++d) {
        int ii = i + d;
        if (ii >= 0 && ii < WD) {
          a  += __expf((float)s[ii*WD + j] - rmax[ii]) / rsum[ii];
          ar += __expf((float)s[j*WD + ii] - cmax[ii]) / csum[ii];
        }
      }
      vl += a  * (__expf(sij - cmax[j]) / csum[j]);
      vr += ar * (__expf(sji - rmax[j]) / rsum[j]);
    }
    Vl[i] = vl; Vr[i] = vr;
  }
  wait_async(); __syncthreads();
  for (int i = tid; i < 64*WD; i += 256) {           // xr -> bufB [w][c] f16
    int c = i >> 8, w2 = i & 255;
    bufB[w2*64 + c] = (half_t)stage[c*WD + w2];
  }
  __syncthreads();

  // ---- out_left: x_lT = M_r2l @ xr, blend with tanh(5 V_l) ----
  v8f z = {0.f,0.f,0.f,0.f,0.f,0.f,0.f,0.f};
  for (int mt = wv; mt < 16; mt += 8) {
    const int i = mt*16 + m;
    const float rm = rmax[i], ri = 1.f / rsum[i];
    v8f acc[4] = {z, z, z, z};
#pragma unroll 1
    for (int kc = 0; kc < 8; ++kc) {
      v16h a;
#pragma unroll
      for (int e = 0; e < 16; ++e) {
        int k = (e < 8) ? (kh*8 + e) : (16 + kh*8 + (e - 8));
        int j = kc*32 + k;
        a[e] = (half_t)(__expf((float)s[i*WD + j] - rm) * ri);
      }
#pragma unroll
      for (int nt = 0; nt < 4; ++nt) {
        const half_t* bp = bufB + (kc*32 + kh*8)*64 + nt*16 + (lane & 15);
        v16h bf;
#pragma unroll
        for (int e = 0; e < 16; ++e) bf[e] = bp[((e < 8) ? e : (8 + e))*64];
        acc[nt] = wmma16(a, bf, acc[nt]);
      }
    }
#pragma unroll
    for (int nt = 0; nt < 4; ++nt) {
      int ch = nt*16 + (lane & 15);
      size_t base = ((size_t)(b*CC + ch)*HT + h)*WD;
      int pb = mt*16 + 8*kh;
#pragma unroll
      for (int r = 0; r < 8; ++r) {
        int px = pb + r;
        float tt = tanhf(5.f * Vl[px]);
        outl[base + px] = xl[rowbase + (size_t)ch*HW + px]*(1.f - tt) + acc[nt][r]*tt;
      }
    }
  }

  // ---- out_right: x_rT = M_l2r @ xl, blend with tanh(5 V_r) ----
  for (int mt = wv; mt < 16; mt += 8) {
    const int i = mt*16 + m;
    const float cm = cmax[i], ci = 1.f / csum[i];
    v8f acc[4] = {z, z, z, z};
#pragma unroll 1
    for (int kc = 0; kc < 8; ++kc) {
      v16h a;
#pragma unroll
      for (int e = 0; e < 16; ++e) {
        int k = (e < 8) ? (kh*8 + e) : (16 + kh*8 + (e - 8));
        int j = kc*32 + k;
        a[e] = (half_t)(__expf((float)s[j*WD + i] - cm) * ci);
      }
#pragma unroll
      for (int nt = 0; nt < 4; ++nt) {
        const half_t* bp = bufA + (kc*32 + kh*8)*64 + nt*16 + (lane & 15);
        v16h bf;
#pragma unroll
        for (int e = 0; e < 16; ++e) bf[e] = bp[((e < 8) ? e : (8 + e))*64];
        acc[nt] = wmma16(a, bf, acc[nt]);
      }
    }
#pragma unroll
    for (int nt = 0; nt < 4; ++nt) {
      int ch = nt*16 + (lane & 15);
      size_t base = ((size_t)(b*CC + ch)*HT + h)*WD;
      int pb = mt*16 + 8*kh;
#pragma unroll
      for (int r = 0; r < 8; ++r) {
        int px = pb + r;
        float tt = tanhf(5.f * Vr[px]);
        outr[base + px] = xr[rowbase + (size_t)ch*HW + px]*(1.f - tt) + acc[nt][r]*tt;
      }
    }
  }
}

// ===========================================================================
// Host launcher
// ===========================================================================
extern "C" void kernel_launch(void* const* d_in, const int* in_sizes, int n_in,
                              void* d_out, int out_size, void* d_ws, size_t ws_size,
                              hipStream_t stream) {
  (void)in_sizes; (void)n_in; (void)out_size; (void)ws_size;
  const float* x_left  = (const float*)d_in[0];
  const float* x_right = (const float*)d_in[1];
  const float* conv1_w = (const float*)d_in[2];
  const float* conv1_b = (const float*)d_in[3];
  const float* conv2_w = (const float*)d_in[4];
  const float* conv2_b = (const float*)d_in[5];
  const float* bn_g    = (const float*)d_in[6];
  const float* bn_b    = (const float*)d_in[7];
  const float* bn_m    = (const float*)d_in[8];
  const float* bn_v    = (const float*)d_in[9];
  const float* rb_w1   = (const float*)d_in[10];
  const float* rb_b1   = (const float*)d_in[11];
  const float* rb_w2   = (const float*)d_in[12];
  const float* rb_b2   = (const float*)d_in[13];
  const float* bq_w    = (const float*)d_in[14];
  const float* bq_b    = (const float*)d_in[15];
  const float* bs_w    = (const float*)d_in[16];
  const float* bs_b    = (const float*)d_in[17];

  const size_t HTENS = (size_t)BB*C2*HT*WD;          // 16,777,216 f16 elems
  char* ws = (char*)d_ws;
  half_t* cat_l = (half_t*)ws;
  half_t* cat_r = cat_l + HTENS;
  half_t* y1_l  = cat_r + HTENS;
  half_t* y1_r  = y1_l  + HTENS;
  half_t* rb_l  = y1_r  + HTENS;
  half_t* rb_r  = rb_l  + HTENS;
  float*  Qb    = (float*)(ws + 6*HTENS*sizeof(half_t));
  float*  Kb    = Qb + (size_t)BB*CC*HT*WD;

  float* out_l = (float*)d_out;
  float* out_r = out_l + (size_t)BB*CC*HT*WD;

  // front: 1x1 conv + concat + BN (per side)
  pam_front<<<2048, 128, 0, stream>>>(x_left,  conv1_w, conv1_b,
                                      bn_g, bn_b, bn_m, bn_v, cat_l);
  pam_front<<<2048, 128, 0, stream>>>(x_right, conv2_w, conv2_b,
                                      bn_g, bn_b, bn_m, bn_v, cat_r);
  // residual block: conv3x3 g4 + lrelu, conv3x3 g4 + residual
  pam_conv3x3g4<<<8192, 128, 0, stream>>>(cat_l, nullptr, y1_l, rb_w1, rb_b1, 1);
  pam_conv3x3g4<<<8192, 128, 0, stream>>>(cat_r, nullptr, y1_r, rb_w1, rb_b1, 1);
  pam_conv3x3g4<<<8192, 128, 0, stream>>>(y1_l, cat_l, rb_l, rb_w2, rb_b2, 0);
  pam_conv3x3g4<<<8192, 128, 0, stream>>>(y1_r, cat_r, rb_r, rb_w2, rb_b2, 0);
  // Q / K projections (1x1 groups=2)
  pam_proj1x1g2<<<4096, 128, 0, stream>>>(rb_l, bq_w, bq_b, Qb);
  pam_proj1x1g2<<<4096, 128, 0, stream>>>(rb_r, bs_w, bs_b, Kb);
  // fused attention: one workgroup per (b,h), 270 KB LDS
  pam_attn<<<512, 256, ATTN_LDS_BYTES, stream>>>(Qb, Kb, x_left, x_right,
                                                 out_l, out_r);
}
